// InformationBottleneck_2070174236952
// MI455X (gfx1250) — compile-verified, matrix operand order checked by reading, not run on hardware
//
#include <hip/hip_runtime.h>

#define N_ROWS 8192
#define DIMD   1024
#define TILE   64
#define KC     32
#define NCHUNK (N_ROWS / KC)
#define NT     (DIMD / TILE)        // 16 tile rows/cols
#define NBLK   (NT * (NT + 1) / 2)  // 136 upper-triangular blocks
#define RS     40                   // padded LDS row stride (halves): 80B, 16B aligned
#define PANEL  (TILE * RS)

typedef __attribute__((ext_vector_type(8)))  __bf16         bf8;
typedef __attribute__((ext_vector_type(16))) __bf16         bf16v;
typedef __attribute__((ext_vector_type(8)))  float          v8f;
typedef __attribute__((ext_vector_type(8)))  unsigned short us8;

#define WMMA_BF16(A, B, C) \
  __builtin_amdgcn_wmma_f32_16x16x32_bf16(false, (A), false, (B), (short)0, (C), false, false)

static __device__ __forceinline__ unsigned short bf16_rne(float f) {
  unsigned int u = __float_as_uint(f);
  u += 0x7FFFu + ((u >> 16) & 1u);
  return (unsigned short)(u >> 16);
}

// Decode linear block id -> upper-triangular (I, J), I <= J. Scalar, <=16 iters.
static __device__ __forceinline__ void tri_decode(int t, int& I, int& J) {
  int i = 0;
  while (t >= NT - i) { t -= NT - i; ++i; }
  I = i;
  J = i + t;
}

// ---------------------------------------------------------------------------
// Pre-pass: split f32 [N][D] into (hi, lo) bf16 and transpose to [D][N].
// ---------------------------------------------------------------------------
__global__ __launch_bounds__(256) void split_transpose_kernel(
    const float* __restrict__ src,
    unsigned short* __restrict__ hiT,
    unsigned short* __restrict__ loT) {
  __shared__ __align__(16) unsigned short sh[TILE * 72];
  __shared__ __align__(16) unsigned short sl[TILE * 72];

  const int t  = threadIdx.x;
  const int nb = blockIdx.x * TILE;
  const int cb = blockIdx.y * TILE;

  {
    const int c  = t & 63;
    const int n0 = t >> 6;
#pragma unroll
    for (int r = 0; r < 16; ++r) {
      const int n = n0 + 4 * r;
      float f  = src[(size_t)(nb + n) * DIMD + cb + c];
      unsigned short hb = bf16_rne(f);
      float hf = __uint_as_float((unsigned int)hb << 16);
      sh[c * 72 + n] = hb;
      sl[c * 72 + n] = bf16_rne(f - hf);
    }
  }
  __syncthreads();
  {
    const int row0 = t >> 3;
    const int g    = t & 7;
#pragma unroll
    for (int h = 0; h < 2; ++h) {
      const int row = row0 + 32 * h;
      us8 vh = *(const us8*)&sh[row * 72 + g * 8];
      us8 vl = *(const us8*)&sl[row * 72 + g * 8];
      *(us8*)&hiT[(size_t)(cb + row) * N_ROWS + nb + g * 8] = vh;
      *(us8*)&loT[(size_t)(cb + row) * N_ROWS + nb + g * 8] = vl;
    }
  }
}

// ---------------------------------------------------------------------------
// Main kernel (pre-split path), upper-triangular tiles only.
// Panels: 0 XiHi 1 XiLo 2 XjHi 3 XjLo 4 YiHi 5 YiLo 6 YjHi 7 YjLo
// ---------------------------------------------------------------------------
__global__ __launch_bounds__(256) void gram_frob_pre_kernel(
    const unsigned short* __restrict__ XhiT, const unsigned short* __restrict__ XloT,
    const unsigned short* __restrict__ YhiT, const unsigned short* __restrict__ YloT,
    float* __restrict__ out) {
  __shared__ __align__(16) unsigned short sm[8 * PANEL];

  const int tid  = threadIdx.x;
  const int lane = tid & 31;
  const int w    = tid >> 5;
  const int cc   = tid >> 2;
  const int gg   = tid & 3;

  int I, J;
  tri_decode(blockIdx.x, I, J);
  const float scale = (I == J) ? 1.0f : 2.0f;

  const size_t iOff = (size_t)(I * TILE) * N_ROWS;
  const size_t jOff = (size_t)(J * TILE) * N_ROWS;

  const unsigned short* srcp[8] = {
    XhiT + iOff, XloT + iOff, XhiT + jOff, XloT + jOff,
    YhiT + iOff, YloT + iOff, YhiT + jOff, YloT + jOff };

  const int tA   = w & 3;
  const int tB0  = (w >> 2) * 2;
  const int tB1  = tB0 + 1;
  const int M    = lane & 15;
  const int half = lane >> 4;

  auto frag = [&](int p, int sub) -> bf16v {
    const unsigned short* base = &sm[p * PANEL + (sub * 16 + M) * RS + half * 8];
    bf8 lo = *(const bf8*)(base);
    bf8 hi = *(const bf8*)(base + 16);
    return __builtin_shufflevector(lo, hi, 0, 1, 2, 3, 4, 5, 6, 7,
                                           8, 9, 10, 11, 12, 13, 14, 15);
  };

  v8f accX0 = {}, accX1 = {}, accY0 = {}, accY1 = {};

  us8 v[8];
  const size_t ldOff = (size_t)cc * N_ROWS + gg * 8;
#pragma unroll
  for (int p = 0; p < 8; ++p) v[p] = *(const us8*)(srcp[p] + ldOff);

  for (int k = 0; k < NCHUNK; ++k) {
#pragma unroll
    for (int p = 0; p < 8; ++p)
      *(us8*)&sm[p * PANEL + cc * RS + gg * 8] = v[p];
    __syncthreads();

    if (k + 1 < NCHUNK) {
      const size_t nb = (size_t)(k + 1) * KC;
#pragma unroll
      for (int p = 0; p < 8; ++p)
        v[p] = *(const us8*)(srcp[p] + ldOff + nb);
    }

    bf16v aXh  = frag(0, tA),  aXl  = frag(1, tA);
    bf16v bXh0 = frag(2, tB0), bXl0 = frag(3, tB0);
    bf16v bXh1 = frag(2, tB1), bXl1 = frag(3, tB1);
    bf16v aYh  = frag(4, tA),  aYl  = frag(5, tA);
    bf16v bYh0 = frag(6, tB0), bYl0 = frag(7, tB0);
    bf16v bYh1 = frag(6, tB1), bYl1 = frag(7, tB1);

    accX0 = WMMA_BF16(aXh, bXh0, accX0);
    accX0 = WMMA_BF16(aXh, bXl0, accX0);
    accX0 = WMMA_BF16(aXl, bXh0, accX0);

    accX1 = WMMA_BF16(aXh, bXh1, accX1);
    accX1 = WMMA_BF16(aXh, bXl1, accX1);
    accX1 = WMMA_BF16(aXl, bXh1, accX1);

    accY0 = WMMA_BF16(aYh, bYh0, accY0);
    accY0 = WMMA_BF16(aYh, bYl0, accY0);
    accY0 = WMMA_BF16(aYl, bYh0, accY0);

    accY1 = WMMA_BF16(aYh, bYh1, accY1);
    accY1 = WMMA_BF16(aYh, bYl1, accY1);
    accY1 = WMMA_BF16(aYl, bYh1, accY1);

    __syncthreads();
  }

  float s = 0.f;
#pragma unroll
  for (int r = 0; r < 8; ++r)
    s += accX0[r] * accY0[r] + accX1[r] * accY1[r];
#pragma unroll
  for (int off = 16; off > 0; off >>= 1)
    s += __shfl_xor(s, off, 32);
  if (lane == 0) atomicAdd(out, s * scale);
}

// ---------------------------------------------------------------------------
// Fallback (fused conversion) path, used only if workspace is too small.
// ---------------------------------------------------------------------------
__global__ __launch_bounds__(256) void gram_frob_fused_kernel(
    const float* __restrict__ X, const float* __restrict__ Y,
    float* __restrict__ out) {
  __shared__ __align__(16) unsigned short sm[8 * PANEL];

  const int tid  = threadIdx.x;
  const int lane = tid & 31;
  const int w    = tid >> 5;
  const int c    = tid & 63;
  const int q    = tid >> 6;

  int I, J;
  tri_decode(blockIdx.x, I, J);
  const float scale = (I == J) ? 1.0f : 2.0f;

  const int iBase = I * TILE;
  const int jBase = J * TILE;
  const float* src[4] = { X + iBase + c, X + jBase + c, Y + iBase + c, Y + jBase + c };

  const int tA   = w & 3;
  const int tB0  = (w >> 2) * 2;
  const int tB1  = tB0 + 1;
  const int M    = lane & 15;
  const int half = lane >> 4;

  auto frag = [&](int p, int sub) -> bf16v {
    const unsigned short* base = &sm[p * PANEL + (sub * 16 + M) * RS + half * 8];
    bf8 lo = *(const bf8*)(base);
    bf8 hi = *(const bf8*)(base + 16);
    return __builtin_shufflevector(lo, hi, 0, 1, 2, 3, 4, 5, 6, 7,
                                           8, 9, 10, 11, 12, 13, 14, 15);
  };

  v8f accX0 = {}, accX1 = {}, accY0 = {}, accY1 = {};

  float v[4][8];
#pragma unroll
  for (int s = 0; s < 4; ++s)
#pragma unroll
    for (int r = 0; r < 8; ++r)
      v[s][r] = src[s][(size_t)(q * 8 + r) * DIMD];

  for (int k = 0; k < NCHUNK; ++k) {
#pragma unroll
    for (int s = 0; s < 4; ++s) {
      us8 h, l;
#pragma unroll
      for (int r = 0; r < 8; ++r) {
        float          f  = v[s][r];
        unsigned short hb = bf16_rne(f);
        float          hf = __uint_as_float((unsigned int)hb << 16);
        h[r] = hb;
        l[r] = bf16_rne(f - hf);
      }
      *(us8*)&sm[(2 * s    ) * PANEL + c * RS + q * 8] = h;
      *(us8*)&sm[(2 * s + 1) * PANEL + c * RS + q * 8] = l;
    }
    __syncthreads();

    if (k + 1 < NCHUNK) {
      const size_t nb = (size_t)(k + 1) * KC;
#pragma unroll
      for (int s = 0; s < 4; ++s)
#pragma unroll
        for (int r = 0; r < 8; ++r)
          v[s][r] = src[s][(nb + (size_t)(q * 8 + r)) * DIMD];
    }

    bf16v aXh  = frag(0, tA),  aXl  = frag(1, tA);
    bf16v bXh0 = frag(2, tB0), bXl0 = frag(3, tB0);
    bf16v bXh1 = frag(2, tB1), bXl1 = frag(3, tB1);
    bf16v aYh  = frag(4, tA),  aYl  = frag(5, tA);
    bf16v bYh0 = frag(6, tB0), bYl0 = frag(7, tB0);
    bf16v bYh1 = frag(6, tB1), bYl1 = frag(7, tB1);

    accX0 = WMMA_BF16(aXh, bXh0, accX0);
    accX0 = WMMA_BF16(aXh, bXl0, accX0);
    accX0 = WMMA_BF16(aXl, bXh0, accX0);
    accX1 = WMMA_BF16(aXh, bXh1, accX1);
    accX1 = WMMA_BF16(aXh, bXl1, accX1);
    accX1 = WMMA_BF16(aXl, bXh1, accX1);
    accY0 = WMMA_BF16(aYh, bYh0, accY0);
    accY0 = WMMA_BF16(aYh, bYl0, accY0);
    accY0 = WMMA_BF16(aYl, bYh0, accY0);
    accY1 = WMMA_BF16(aYh, bYh1, accY1);
    accY1 = WMMA_BF16(aYh, bYl1, accY1);
    accY1 = WMMA_BF16(aYl, bYh1, accY1);

    __syncthreads();
  }

  float s = 0.f;
#pragma unroll
  for (int r = 0; r < 8; ++r)
    s += accX0[r] * accY0[r] + accX1[r] * accY1[r];
#pragma unroll
  for (int off = 16; off > 0; off >>= 1)
    s += __shfl_xor(s, off, 32);
  if (lane == 0) atomicAdd(out, s * scale);
}

extern "C" void kernel_launch(void* const* d_in, const int* in_sizes, int n_in,
                              void* d_out, int out_size, void* d_ws, size_t ws_size,
                              hipStream_t stream) {
  (void)in_sizes; (void)n_in; (void)out_size;
  const float* X = (const float*)d_in[0];
  const float* Y = (const float*)d_in[1];
  float* out = (float*)d_out;

  hipMemsetAsync(out, 0, sizeof(float), stream);

  const size_t arr  = (size_t)DIMD * N_ROWS;
  const size_t need = 4 * arr * sizeof(unsigned short);  // 64 MB

  if (ws_size >= need) {
    unsigned short* XhiT = (unsigned short*)d_ws;
    unsigned short* XloT = XhiT + arr;
    unsigned short* YhiT = XloT + arr;
    unsigned short* YloT = YhiT + arr;

    dim3 tgrid(N_ROWS / TILE, DIMD / TILE);
    split_transpose_kernel<<<tgrid, 256, 0, stream>>>(X, XhiT, XloT);
    split_transpose_kernel<<<tgrid, 256, 0, stream>>>(Y, YhiT, YloT);
    gram_frob_pre_kernel<<<NBLK, 256, 0, stream>>>(XhiT, XloT, YhiT, YloT, out);
  } else {
    gram_frob_fused_kernel<<<NBLK, 256, 0, stream>>>(X, Y, out);
  }
}
